// Critic_18124761989519
// MI455X (gfx1250) — matrix operand order repaired and checked by reading, not compiled, not run on hardware
//
#include <hip/hip_runtime.h>
#include <hip/hip_bf16.h>

// ---------------------------------------------------------------------------
// MI455X (gfx1250, wave32) implementation.
// TextCNN convs + LSTM gate GEMMs + linearX/Y all mapped to
// v_wmma_f32_16x16x32_f16 (f16 operands, f32 accumulate).
// K-outer multi-accumulator structure: B fragments loaded once per K step,
// 3-4 independent WMMA chains per wave to cover XDL latency. Conv K-loop is
// split into tap(j) x inner(i0) so all addressing is additive (no div/mod in
// the hot loop).
// Attention in the reference softmaxes over a size-1 axis -> weights == 1,
// so ctx = sum over days of h_t (computed as a running sum).
// ---------------------------------------------------------------------------

typedef _Float16 v8h  __attribute__((ext_vector_type(8)));
typedef _Float16 v16h __attribute__((ext_vector_type(16)));
typedef float    v8f  __attribute__((ext_vector_type(8)));

// ---- WMMA fragment builders (ISA 7.12.2 layouts, wave32) -------------------
// A (16x32 f16, MxK): lanes 0-15 -> M=lane, halves 0..7 = K[kb+0..7],
//                     halves 8..15 = K[kb+16..23];
//                     lanes 16-31 -> M=lane-16, halves 0..7 = K[kb+8..15],
//                     halves 8..15 = K[kb+24..31].
__device__ inline v16h load_a_frag(const _Float16* row, int kb, int lane) {
  int o0 = kb + ((lane >> 4) << 3);          // +0 or +8
  v8h lo = *(const v8h*)(row + o0);
  v8h hi = *(const v8h*)(row + o0 + 16);
  v16h r;
#pragma unroll
  for (int i = 0; i < 8; ++i) { r[i] = lo[i]; r[i + 8] = hi[i]; }
  return r;
}
// B (32x16 f16, KxN): lane L -> N = L&15; same K order as A:
// lanes 0-15 halves 0..7 = K[kb+0..7], 8..15 = K[kb+16..23]; lanes 16-31 +8.
__device__ inline v16h load_b_frag(const _Float16* row, int kb, int lane) {
  int o0 = kb + ((lane >> 4) << 3);          // +0 or +8
  v8h lo = *(const v8h*)(row + o0);
  v8h hi = *(const v8h*)(row + o0 + 16);
  v16h r;
#pragma unroll
  for (int i = 0; i < 8; ++i) { r[i] = lo[i]; r[i + 8] = hi[i]; }
  return r;
}

__device__ inline float sigmf(float x) { return 1.f / (1.f + __expf(-x)); }

// XOR-16 max across the two lane halves via ds_swizzle (group-of-32, xor=0x10)
__device__ inline float xor16_max(float v) {
  int iv = __builtin_bit_cast(int, v);
  int ov = __builtin_amdgcn_ds_swizzle(iv, 0x401F);
  return fmaxf(v, __builtin_bit_cast(float, ov));
}

// ---- constants -------------------------------------------------------------
#define S_N   10
#define BS_N  16
#define D_N   16
#define N_N   48
#define E_N   300
#define EP    320      // E padded to multiple of 32
#define OCP   112      // OC=100 padded to multiple of 16
#define H_N   128
#define G4H   512

// ---------------------------------------------------------------------------
// Weight packing kernels (run every launch; deterministic)
// ---------------------------------------------------------------------------
__global__ void pack_conv_kernel(const float* __restrict__ W, _Float16* __restrict__ dst, int k) {
  int oc = blockIdx.x;          // 0..111
  int s  = blockIdx.y;          // 0..9
  int Kt = k * EP;
  _Float16* drow = dst + ((size_t)(s * OCP + oc)) * Kt;
  for (int kk = threadIdx.x; kk < Kt; kk += 256) {
    int j = kk / EP, i = kk - j * EP;
    float v = 0.f;
    if (oc < 100 && i < E_N) v = W[((size_t)(s * 100 + oc) * E_N + i) * k + j];
    drow[kk] = (_Float16)v;
  }
}

// [S,R,C] f32 -> [S,R,Cp] f16 zero-padded row-major
__global__ void pack_mat_kernel(const float* __restrict__ src, _Float16* __restrict__ dst,
                                int R, int C, int Cp) {
  int r = blockIdx.x, s = blockIdx.y;
  _Float16* drow = dst + ((size_t)(s * R + r)) * Cp;
  const float* srow = src + ((size_t)(s * R + r)) * C;
  for (int c = threadIdx.x; c < Cp; c += 256)
    drow[c] = (_Float16)((c < C) ? srow[c] : 0.f);
}

// ---------------------------------------------------------------------------
// TextCNN: one block (7 waves) per (item, stock). Implicit-GEMM conv + maxpool.
// tap(j)-outer / i0-inner: one B fragment feeds 3 M-tile accumulators
// (3-way WMMA ILP, 3x less B traffic), all addresses additive.
// ---------------------------------------------------------------------------
__global__ __launch_bounds__(224) void textcnn_kernel(
    const float* __restrict__ news,
    const float* __restrict__ bc3, const float* __restrict__ bc4, const float* __restrict__ bc5,
    const _Float16* __restrict__ B3, const _Float16* __restrict__ B4, const _Float16* __restrict__ B5,
    _Float16* __restrict__ feats) {
  __shared__ _Float16 sx[52 * EP];           // time-major x, zero padded
  int item = blockIdx.x;                     // bs*D + d
  int s  = blockIdx.y;
  int bs = item >> 4;
  int d  = item & 15;
  const float* src = news + (((size_t)(bs * S_N + s) * D_N + d) * N_N) * E_N;
  for (int idx = threadIdx.x; idx < 52 * EP; idx += 224) {
    int t = idx / EP, i = idx - t * EP;
    float v = (t < N_N && i < E_N) ? src[t * E_N + i] : 0.f;
    sx[idx] = (_Float16)v;
  }
  _Float16* frow = feats + ((size_t)(s * D_N + d) * BS_N + bs) * EP;
  if (threadIdx.x < EP - E_N) frow[E_N + threadIdx.x] = (_Float16)0.f;  // K pad
  __syncthreads();

  int lane = threadIdx.x & 31;
  int nt   = threadIdx.x >> 5;               // 0..6 : OC tile
  int ncol = lane & 15;
  int oc   = nt * 16 + ncol;

  const _Float16* Bp[3]  = { B3, B4, B5 };
  const float*   bias[3] = { bc3, bc4, bc5 };
#pragma unroll
  for (int ki = 0; ki < 3; ++ki) {
    const int k  = 3 + ki;
    const int Kt = k * EP;
    const int tvalid = N_N - k + 1;
    const _Float16* bj = Bp[ki] + ((size_t)(s * OCP + oc)) * Kt;  // advances by EP per tap

    v8f acc[3];
#pragma unroll
    for (int mt = 0; mt < 3; ++mt) acc[mt] = (v8f){};

    for (int j = 0; j < k; ++j) {
      const _Float16* aj = sx + (ncol + j) * EP;   // im2col row base for tap j
#pragma unroll 5
      for (int i0 = 0; i0 < EP; i0 += 32) {
        __builtin_prefetch(bj + i0 + 256, 0, 1);   // B stream, few frags ahead
        v16h b = load_b_frag(bj, i0, lane);
#pragma unroll
        for (int mt = 0; mt < 3; ++mt) {
          v16h a = load_a_frag(aj + mt * (16 * EP), i0, lane);
          acc[mt] = __builtin_amdgcn_wmma_f32_16x16x32_f16(false, a, false, b,
                                                           (short)0, acc[mt], false, false);
        }
      }
      bj += EP;
    }
    float maxv = -1e30f;
    int mhalf = (lane < 16) ? 0 : 8;
#pragma unroll
    for (int mt = 0; mt < 3; ++mt) {
      int mbase = mt * 16 + mhalf;
#pragma unroll
      for (int r = 0; r < 8; ++r) {
        int t = mbase + r;
        if (t < tvalid) maxv = fmaxf(maxv, acc[mt][r]);
      }
    }
    maxv = xor16_max(maxv);                  // merge the two M halves
    if (lane < 16 && oc < 100)
      frow[ki * 100 + oc] = (_Float16)(maxv + bias[ki][s * 100 + oc]);
  }
}

// ---------------------------------------------------------------------------
// Per-stock LSTM (+attention-sum + linearX + linearY). One 8-wave block/stock.
// Gate GEMM: K-outer with 4 N-tile accumulators per wave (A loaded once/K).
// ---------------------------------------------------------------------------
__global__ __launch_bounds__(256) void lstm_kernel(
    const _Float16* __restrict__ feats,
    const _Float16* __restrict__ Wih, const _Float16* __restrict__ Whh,
    const float* __restrict__ b_ih, const float* __restrict__ b_hh,
    const _Float16* __restrict__ Wxp, const float* __restrict__ bx,
    const _Float16* __restrict__ Wyp, const float* __restrict__ by,
    float* __restrict__ text) {
  __shared__ float    gates[BS_N * G4H];     // 32 KB
  __shared__ float    cbuf[BS_N * H_N];
  __shared__ float    ctx [BS_N * H_N];
  __shared__ _Float16 hbuf[BS_N * H_N];
  __shared__ _Float16 abuf[BS_N * 256];
  __shared__ _Float16 zbuf[BS_N * H_N];
  int s = blockIdx.x;
  int tid = threadIdx.x, lane = tid & 31, w = tid >> 5;
  int ncol = lane & 15;
  for (int i = tid; i < BS_N * H_N; i += 256) {
    cbuf[i] = 0.f; ctx[i] = 0.f; hbuf[i] = (_Float16)0.f;
  }
  __syncthreads();

  for (int d = 0; d < D_N; ++d) {
    const _Float16* xbase = feats + ((size_t)(s * D_N + d) * BS_N) * EP;
    const _Float16* arow  = xbase + (size_t)ncol * EP;             // A row = bs
    const _Float16* hrow  = hbuf + ncol * H_N;

    v8f acc[4];
    const _Float16* browx[4];
    const _Float16* browh[4];
#pragma unroll
    for (int q = 0; q < 4; ++q) {
      int n = (w * 4 + q) * 16 + ncol;       // gate column (0..511)
      acc[q]   = (v8f){};
      browx[q] = Wih + ((size_t)(s * G4H + n)) * EP;
      browh[q] = Whh + ((size_t)(s * G4H + n)) * H_N;
    }
    for (int kb = 0; kb < EP; kb += 32) {    // x @ W_ih^T
      v16h a = load_a_frag(arow, kb, lane);
#pragma unroll
      for (int q = 0; q < 4; ++q) {
        v16h b = load_b_frag(browx[q], kb, lane);
        acc[q] = __builtin_amdgcn_wmma_f32_16x16x32_f16(false, a, false, b,
                                                        (short)0, acc[q], false, false);
      }
    }
#pragma unroll
    for (int kb = 0; kb < H_N; kb += 32) {   // h @ W_hh^T
      v16h a = load_a_frag(hrow, kb, lane);
#pragma unroll
      for (int q = 0; q < 4; ++q) {
        v16h b = load_b_frag(browh[q], kb, lane);
        acc[q] = __builtin_amdgcn_wmma_f32_16x16x32_f16(false, a, false, b,
                                                        (short)0, acc[q], false, false);
      }
    }
    int mb = (lane < 16) ? 0 : 8;
#pragma unroll
    for (int q = 0; q < 4; ++q) {
      int n = (w * 4 + q) * 16 + ncol;
      float bsum = b_ih[s * G4H + n] + b_hh[s * G4H + n];
#pragma unroll
      for (int r = 0; r < 8; ++r) gates[(mb + r) * G4H + n] = acc[q][r] + bsum;
    }
    __syncthreads();
    for (int idx = tid; idx < BS_N * H_N; idx += 256) {
      int b = idx >> 7, u = idx & 127;
      float gi = gates[b * G4H + u];
      float gf = gates[b * G4H + 128 + u];
      float gg = gates[b * G4H + 256 + u];
      float go = gates[b * G4H + 384 + u];
      float c = sigmf(gf) * cbuf[idx] + sigmf(gi) * tanhf(gg);
      float h = sigmf(go) * tanhf(c);
      cbuf[idx] = c;
      hbuf[idx] = (_Float16)h;
      ctx[idx] += h;                          // faithful "softmax over size-1 axis"
    }
    __syncthreads();
  }

  // lstm_out = [hn, ctx]  (16 x 256), f16 for WMMA
  for (int idx = tid; idx < BS_N * 256; idx += 256) {
    int b = idx >> 8, c = idx & 255;
    abuf[idx] = (c < H_N) ? hbuf[b * H_N + c] : (_Float16)ctx[b * H_N + (c - H_N)];
  }
  __syncthreads();

  // linearX: N=128 -> 8 tiles, one per wave
  {
    int n = w * 16 + ncol;
    const _Float16* arow = abuf + ncol * 256;
    const _Float16* brow = Wxp + ((size_t)(s * H_N + n)) * 256;
    v8f acc = {};
#pragma unroll
    for (int kb = 0; kb < 256; kb += 32) {
      v16h a = load_a_frag(arow, kb, lane);
      v16h b = load_b_frag(brow, kb, lane);
      acc = __builtin_amdgcn_wmma_f32_16x16x32_f16(false, a, false, b,
                                                   (short)0, acc, false, false);
    }
    float bb = bx[s * H_N + n];
    int mb = (lane < 16) ? 0 : 8;
#pragma unroll
    for (int r = 0; r < 8; ++r)
      zbuf[(mb + r) * H_N + n] = (_Float16)fmaxf(acc[r] + bb, 0.f);
  }
  __syncthreads();

  // linearY: N=64 -> 4 tiles on waves 0..3 (wave-uniform branch; EXEC all-1s)
  if (w < 4) {
    int n = w * 16 + ncol;
    const _Float16* arow = zbuf + ncol * H_N;
    const _Float16* brow = Wyp + ((size_t)(s * 64 + n)) * H_N;
    v8f acc = {};
#pragma unroll
    for (int kb = 0; kb < H_N; kb += 32) {
      v16h a = load_a_frag(arow, kb, lane);
      v16h b = load_b_frag(brow, kb, lane);
      acc = __builtin_amdgcn_wmma_f32_16x16x32_f16(false, a, false, b,
                                                   (short)0, acc, false, false);
    }
    float bb = by[s * 64 + n];
    int mb = (lane < 16) ? 0 : 8;
#pragma unroll
    for (int r = 0; r < 8; ++r)
      text[(size_t)(mb + r) * 640 + s * 64 + n] = fmaxf(acc[r] + bb, 0.f);
  }
}

// ---------------------------------------------------------------------------
// Head: tiny f32 scalar math, one block.
// ---------------------------------------------------------------------------
__global__ __launch_bounds__(256) void head_kernel(
    const float* __restrict__ text, const float* __restrict__ stock_feats,
    const float* __restrict__ action,
    const float* __restrict__ W1, const float* __restrict__ b1,
    const float* __restrict__ W2, const float* __restrict__ b2,
    const float* __restrict__ Wc, const float* __restrict__ bc,
    const float* __restrict__ Wca, const float* __restrict__ bca,
    const float* __restrict__ Wa1, const float* __restrict__ ba1,
    const float* __restrict__ Wa2, const float* __restrict__ ba2,
    float* __restrict__ out) {
  __shared__ float hid1[16 * 24];
  __shared__ float sf[16 * 16];
  __shared__ float comb[16 * 32];
  __shared__ float ca[16 * 16];
  __shared__ float a1[16 * 32];
  int tid = threadIdx.x;
  for (int idx = tid; idx < 16 * 24; idx += 256) {
    int b = idx / 24, j = idx % 24;
    float acc = b1[j];
    for (int c = 0; c < 100; ++c) acc += stock_feats[b * 100 + c] * W1[j * 100 + c];
    hid1[idx] = fmaxf(acc, 0.f);
  }
  __syncthreads();
  for (int idx = tid; idx < 16 * 16; idx += 256) {
    int b = idx / 16, j = idx % 16;
    float acc = b2[j];
    for (int c = 0; c < 24; ++c) acc += hid1[b * 24 + c] * W2[j * 24 + c];
    sf[idx] = acc;
  }
  __syncthreads();
  for (int idx = tid; idx < 16 * 32; idx += 256) {
    int b = idx / 32, j = idx % 32;
    float acc = bc[j];
    for (int c = 0; c < 640; ++c) acc += text[b * 640 + c] * Wc[j * 656 + c];
    for (int c = 0; c < 16; ++c)  acc += sf[b * 16 + c] * Wc[j * 656 + 640 + c];
    comb[idx] = acc;
  }
  __syncthreads();
  for (int idx = tid; idx < 16 * 16; idx += 256) {
    int b = idx / 16, j = idx % 16;
    float acc = bca[j];
    for (int c = 0; c < 32; ++c) acc += comb[b * 32 + c] * Wca[j * 42 + c];
    for (int c = 0; c < 10; ++c) acc += action[b * 10 + c] * Wca[j * 42 + 32 + c];
    ca[idx] = acc;
  }
  __syncthreads();
  for (int idx = tid; idx < 16 * 32; idx += 256) {
    int b = idx / 32, j = idx % 32;
    float acc = ba1[j];
    for (int c = 0; c < 16; ++c) acc += ca[b * 16 + c] * Wa1[j * 16 + c];
    a1[idx] = fmaxf(acc, 0.f);
  }
  __syncthreads();
  if (tid < 16) {
    float acc = ba2[0];
    for (int c = 0; c < 32; ++c) acc += a1[tid * 32 + c] * Wa2[c];
    out[tid] = acc;
  }
}

// ---------------------------------------------------------------------------
extern "C" void kernel_launch(void* const* d_in, const int* in_sizes, int n_in,
                              void* d_out, int out_size, void* d_ws, size_t ws_size,
                              hipStream_t stream) {
  (void)in_sizes; (void)n_in; (void)out_size; (void)ws_size;
  const float* news        = (const float*)d_in[0];
  const float* stock_feats = (const float*)d_in[1];
  const float* action      = (const float*)d_in[2];
  const float* Wc3 = (const float*)d_in[3];  const float* bc3 = (const float*)d_in[4];
  const float* Wc4 = (const float*)d_in[5];  const float* bc4 = (const float*)d_in[6];
  const float* Wc5 = (const float*)d_in[7];  const float* bc5 = (const float*)d_in[8];
  const float* W_ih = (const float*)d_in[9];  const float* W_hh = (const float*)d_in[10];
  const float* b_ih = (const float*)d_in[11]; const float* b_hh = (const float*)d_in[12];
  const float* Wx = (const float*)d_in[13];   const float* bx = (const float*)d_in[14];
  const float* Wy = (const float*)d_in[15];   const float* by = (const float*)d_in[16];
  const float* W1 = (const float*)d_in[17];   const float* b1 = (const float*)d_in[18];
  const float* W2 = (const float*)d_in[19];   const float* b2 = (const float*)d_in[20];
  const float* Wc = (const float*)d_in[21];   const float* bc = (const float*)d_in[22];
  const float* Wca = (const float*)d_in[23];  const float* bca = (const float*)d_in[24];
  const float* Wa1 = (const float*)d_in[25];  const float* ba1 = (const float*)d_in[26];
  const float* Wa2 = (const float*)d_in[27];  const float* ba2 = (const float*)d_in[28];
  float* out = (float*)d_out;

  // ---- workspace layout (bytes, 256-aligned) ----
  auto align256 = [](size_t x) { return (x + 255) & ~(size_t)255; };
  char* ws = (char*)d_ws;
  size_t off = 0;
  float* text = (float*)(ws + off);             off = align256(off + (size_t)BS_N * 640 * 4);
  _Float16* feats = (_Float16*)(ws + off);      off = align256(off + (size_t)S_N * D_N * BS_N * EP * 2);
  _Float16* B3 = (_Float16*)(ws + off);         off = align256(off + (size_t)S_N * OCP * (3 * EP) * 2);
  _Float16* B4 = (_Float16*)(ws + off);         off = align256(off + (size_t)S_N * OCP * (4 * EP) * 2);
  _Float16* B5 = (_Float16*)(ws + off);         off = align256(off + (size_t)S_N * OCP * (5 * EP) * 2);
  _Float16* Wihp = (_Float16*)(ws + off);       off = align256(off + (size_t)S_N * G4H * EP * 2);
  _Float16* Whhp = (_Float16*)(ws + off);       off = align256(off + (size_t)S_N * G4H * H_N * 2);
  _Float16* Wxp = (_Float16*)(ws + off);        off = align256(off + (size_t)S_N * H_N * 256 * 2);
  _Float16* Wyp = (_Float16*)(ws + off);        off = align256(off + (size_t)S_N * 64 * H_N * 2);

  // ---- pack weights to f16 (deterministic, every call) ----
  pack_conv_kernel<<<dim3(OCP, S_N), 256, 0, stream>>>(Wc3, B3, 3);
  pack_conv_kernel<<<dim3(OCP, S_N), 256, 0, stream>>>(Wc4, B4, 4);
  pack_conv_kernel<<<dim3(OCP, S_N), 256, 0, stream>>>(Wc5, B5, 5);
  pack_mat_kernel<<<dim3(G4H, S_N), 256, 0, stream>>>(W_ih, Wihp, G4H, E_N, EP);
  pack_mat_kernel<<<dim3(G4H, S_N), 256, 0, stream>>>(W_hh, Whhp, G4H, H_N, H_N);
  pack_mat_kernel<<<dim3(H_N, S_N), 256, 0, stream>>>(Wx, Wxp, H_N, 256, 256);
  pack_mat_kernel<<<dim3(64,  S_N), 256, 0, stream>>>(Wy, Wyp, 64, H_N, H_N);

  // ---- TextCNN (bulk of the 83 GFLOP, WMMA implicit GEMM) ----
  textcnn_kernel<<<dim3(BS_N * D_N, S_N), 224, 0, stream>>>(
      news, bc3, bc4, bc5, B3, B4, B5, feats);

  // ---- per-stock LSTM + linears (WMMA) ----
  lstm_kernel<<<S_N, 256, 0, stream>>>(feats, Wihp, Whhp, b_ih, b_hh,
                                       Wxp, bx, Wyp, by, text);

  // ---- tiny head ----
  head_kernel<<<1, 256, 0, stream>>>(text, stock_feats, action,
                                     W1, b1, W2, b2, Wc, bc, Wca, bca,
                                     Wa1, ba1, Wa2, ba2, out);
}
